// MultiHeadAttention_86638080295429
// MI455X (gfx1250) — matrix operand order repaired and checked by reading, compiled, not run on hardware
//
#include <hip/hip_runtime.h>
#include <hip/hip_bf16.h>

// MHA forward, MI455X (gfx1250, wave32, WMMA + Tensor Data Mover).
// Pipeline: cvt(f32->bf16) -> 3x GEMM (QKV, scatter to BHTD / BHDT) ->
//           flash attention (WMMA bf16, fp32 softmax) -> GEMM (out proj, f32).
// All LDS tile staging is done by the TDM (tensor_load_to_lds, TENSORcnt),
// double-buffered so the DMA overlaps the WMMA stream.

#define B_  2
#define T_  2048
#define C_  1024
#define H_  16
#define D_  64
#define M_  (B_ * T_)      // 4096 rows for the projection GEMMs

typedef __attribute__((ext_vector_type(16))) __bf16 bf16x16;
typedef __attribute__((ext_vector_type(8)))  __bf16 bf16x8;
typedef __attribute__((ext_vector_type(4)))  __bf16 bf16x4;
typedef __attribute__((ext_vector_type(8)))  float  f32x8;
typedef __attribute__((ext_vector_type(4)))  unsigned int u32x4;
typedef __attribute__((ext_vector_type(8)))  int i32x8;
typedef __attribute__((ext_vector_type(4)))  int i32x4;

static __device__ __forceinline__ bf16x16 join8(bf16x8 lo, bf16x8 hi) {
  return __builtin_shufflevector(lo, hi, 0,1,2,3,4,5,6,7,8,9,10,11,12,13,14,15);
}
static __device__ __forceinline__ f32x8 wmma_bf16(bf16x16 a, bf16x16 b, f32x8 c) {
  // D = A(16x32 bf16) * B(32x16 bf16) + C(16x16 f32)
  return __builtin_amdgcn_wmma_f32_16x16x32_bf16(false, a, false, b, (short)0, c,
                                                 false, false);
}
static __device__ __forceinline__ float rmax16(float v) {
  v = fmaxf(v, __shfl_xor(v, 1, 32));
  v = fmaxf(v, __shfl_xor(v, 2, 32));
  v = fmaxf(v, __shfl_xor(v, 4, 32));
  v = fmaxf(v, __shfl_xor(v, 8, 32));
  return v;
}
static __device__ __forceinline__ float rsum16(float v) {
  v += __shfl_xor(v, 1, 32);
  v += __shfl_xor(v, 2, 32);
  v += __shfl_xor(v, 4, 32);
  v += __shfl_xor(v, 8, 32);
  return v;
}

// ------------------------------------------------------ TDM 2D tile -> LDS
// D# per CDNA5 ISA ch.8: group0 {count=1, lds_addr, global_addr, type=2},
// group1 {data_size=2B, tensor_dim0/1, tile_dim0/1, tensor_dim0_stride}.
// Dims/strides in elements (data_size units); addresses in bytes.
static __device__ __forceinline__ void tdm_load_2d(
    unsigned lds_addr, const void* gptr,
    unsigned tile_d0, unsigned tile_d1,
    unsigned tensor_d0, unsigned tensor_d1,
    unsigned long long stride0) {
  unsigned long long ga = (unsigned long long)gptr;
  u32x4 g0;
  g0.x = 1u;                                    // count=1, no gather/iterate
  g0.y = lds_addr;                              // LDS byte address
  g0.z = (unsigned)ga;                          // global_addr[31:0]
  g0.w = (unsigned)((ga >> 32) & 0x1FFFFFFull)  // global_addr[56:32]
       | (2u << 30);                            // type=2 ("image")
  i32x8 g1;
  g1[0] = (int)(1u << 16);                                  // data_size=1 -> 2B
  g1[1] = (int)((tensor_d0 & 0xFFFFu) << 16);               // tensor_dim0 lo
  g1[2] = (int)((tensor_d0 >> 16) | ((tensor_d1 & 0xFFFFu) << 16));
  g1[3] = (int)(((tensor_d1 >> 16) & 0xFFFFu) | (tile_d0 << 16));
  g1[4] = (int)(tile_d1 & 0xFFFFu);                         // tile_dim2=0
  g1[5] = (int)(unsigned)stride0;                           // stride0[31:0]
  g1[6] = (int)((unsigned)(stride0 >> 32) & 0xFFFFu);       // stride0[47:32]
  g1[7] = 0;
  i32x4 z4 = {0, 0, 0, 0};                                  // 2D: groups 2/3 off
#if __has_include(<hip/amd_detail/amd_gfx1250_TDM.h>)
  i32x8 z8 = {};
  __builtin_amdgcn_tensor_load_to_lds(g0, g1, z4, z4, z8, 0);
#else
  __builtin_amdgcn_tensor_load_to_lds(g0, g1, z4, z4, 0);
#endif
}
static __device__ __forceinline__ unsigned lds_off(const void* p) {
  return (unsigned)(unsigned long long)p;       // low 32 bits = LDS byte offset
}

// ---------------------------------------------------------------- convert
__global__ __launch_bounds__(256)
void mha_cvt_bf16(const float* __restrict__ s, __bf16* __restrict__ d, int n) {
  int i = (blockIdx.x * 256 + threadIdx.x) * 4;
  if (i + 3 < n) {
    const float4 v = *(const float4*)(s + i);
    bf16x4 o;
    o.x = (__bf16)v.x; o.y = (__bf16)v.y; o.z = (__bf16)v.z; o.w = (__bf16)v.w;
    *(bf16x4*)(d + i) = o;
  }
}

// ---------------------------------------------------------------- GEMM
// D = A(MxK, row major bf16) * W(NxK, row major bf16)^T
// block tile 128x128, 8 waves (4x2), wave tile 32x64, K-step 32.
// mode 0: bf16 scatter -> q/k layout [(b*H+h)][t][d]
// mode 1: bf16 scatter -> vT layout  [(b*H+h)][d][t]
// mode 2: f32 row-major -> d_out
__global__ __launch_bounds__(256, 2)
void mha_gemm_wmma(const __bf16* __restrict__ A, const __bf16* __restrict__ W,
                   void* __restrict__ Dst, int M, int N, int K, int mode) {
  __shared__ alignas(128) __bf16 As[2][128][32];
  __shared__ alignas(128) __bf16 Bs[2][128][32];

  const int tid  = threadIdx.x;
  const int lane = tid & 31, wave = tid >> 5;
  const int wm   = wave >> 1, wn = wave & 1;     // 4x2 wave grid
  const int lm   = lane & 15, half = lane >> 4;
  const int b0   = half ? 16 : 0;                // A-fragment K-group select
  const int m0   = blockIdx.y * 128, n0 = blockIdx.x * 128;

  f32x8 zf = {};
  f32x8 acc[2][4] = {zf, zf, zf, zf, zf, zf, zf, zf};

  const int steps = K / 32;
  if (wave == 0) {  // prime buffer 0 via TDM
    tdm_load_2d(lds_off(&As[0][0][0]), A + (size_t)m0 * K, 32, 128, K, M, K);
    tdm_load_2d(lds_off(&Bs[0][0][0]), W + (size_t)n0 * K, 32, 128, K, N, K);
    __builtin_amdgcn_s_wait_tensorcnt(0);
  }
  __syncthreads();

  for (int it = 0; it < steps; ++it) {
    const int cur = it & 1, nxt = cur ^ 1;
    if (it + 1 < steps && wave == 0) {  // async fill of next buffer (overlaps WMMA)
      const int k0 = (it + 1) * 32;
      tdm_load_2d(lds_off(&As[nxt][0][0]), A + (size_t)m0 * K + k0, 32, 128, K, M, K);
      tdm_load_2d(lds_off(&Bs[nxt][0][0]), W + (size_t)n0 * K + k0, 32, 128, K, N, K);
    }
    bf16x16 af[2], bfr[4];
#pragma unroll
    for (int sm = 0; sm < 2; ++sm) {
      int mr = wm * 32 + sm * 16 + lm;
      af[sm] = join8(*(const bf16x8*)&As[cur][mr][b0],
                     *(const bf16x8*)&As[cur][mr][b0 + 8]);
    }
#pragma unroll
    for (int sn = 0; sn < 4; ++sn) {
      int nr = wn * 64 + sn * 16 + lm;
      bfr[sn] = *(const bf16x16*)&Bs[cur][nr][half * 16];
    }
#pragma unroll
    for (int sm = 0; sm < 2; ++sm)
#pragma unroll
      for (int sn = 0; sn < 4; ++sn)
        acc[sm][sn] = wmma_bf16(af[sm], bfr[sn], acc[sm][sn]);
    if (wave == 0) __builtin_amdgcn_s_wait_tensorcnt(0);
    __syncthreads();
  }

  // epilogue: C layout -> lane holds col lm, rows j + 8*half per VGPR j
#pragma unroll
  for (int sm = 0; sm < 2; ++sm) {
#pragma unroll
    for (int sn = 0; sn < 4; ++sn) {
#pragma unroll
      for (int j = 0; j < 8; ++j) {
        float v = acc[sm][sn][j];
        int gm = m0 + wm * 32 + sm * 16 + j + 8 * half;
        int gn = n0 + wn * 64 + sn * 16 + lm;
        if (mode == 2) {
          ((float*)Dst)[(size_t)gm * N + gn] = v;
        } else {
          int bb = gm >> 11, t = gm & (T_ - 1);
          int h  = gn >> 6,  d = gn & (D_ - 1);
          size_t idx = (mode == 0)
              ? ((size_t)((bb * H_ + h) * T_ + t) * D_ + d)
              : ((size_t)((bb * H_ + h) * D_ + d) * T_ + t);
          ((__bf16*)Dst)[idx] = (__bf16)v;
        }
      }
    }
  }
}

// ---------------------------------------------------------------- flash attn
// grid: (T/64, B*H); block: 128 threads = 4 waves; wave owns 16 query rows.
// K/V^T tiles double-buffered via TDM.
__global__ __launch_bounds__(128, 2)
void mha_flash_wmma(const __bf16* __restrict__ q, const __bf16* __restrict__ k,
                    const __bf16* __restrict__ vT, __bf16* __restrict__ attn) {
  __shared__ alignas(128) __bf16 kbuf[2][32][64];   // 32 keys x 64 d
  __shared__ alignas(128) __bf16 vbuf[2][64][32];   // 64 d x 32 keys (V^T)
  __shared__ alignas(64)  __bf16 pbuf[4][16][32];   // per-wave P re-stripe

  const int tid  = threadIdx.x;
  const int lane = tid & 31, wave = tid >> 5;
  const int lm   = lane & 15, half = lane >> 4;
  const int b0   = half ? 16 : 0;
  const int qb   = blockIdx.x;               // 64-row query block
  const int bh   = blockIdx.y;               // b*H + h
  const int b    = bh >> 4, h = bh & 15;
  const int tq0  = qb * 64 + wave * 16;      // wave's first query row
  const size_t base = (size_t)bh * T_ * D_;
  const float NEG = -__builtin_inff();

  // Q fragments (16 rows x 64 d), kept in VGPRs for the whole loop
  bf16x16 qf[2];
#pragma unroll
  for (int kk = 0; kk < 2; ++kk) {
    const __bf16* qp = q + base + (size_t)(tq0 + lm) * D_ + kk * 32 + b0;
    qf[kk] = join8(*(const bf16x8*)qp, *(const bf16x8*)(qp + 8));
  }

  f32x8 zf = {};
  f32x8 ao[4] = {zf, zf, zf, zf};
  float mrow[8], lrow[8];
#pragma unroll
  for (int j = 0; j < 8; ++j) { mrow[j] = NEG; lrow[j] = 0.f; }

  const int kbmax = (qb * 64 + 63) >> 5;     // causal: last key block
  if (wave == 0) {  // prime buffer 0
    tdm_load_2d(lds_off(&kbuf[0][0][0]), k + base, 64, 32, 64, T_, 64);
    tdm_load_2d(lds_off(&vbuf[0][0][0]), vT + base, 32, 64, T_, 64, T_);
    __builtin_amdgcn_s_wait_tensorcnt(0);
  }
  __syncthreads();

  for (int kb = 0; kb <= kbmax; ++kb) {
    const int cur = kb & 1, nxt = cur ^ 1;
    if (kb + 1 <= kbmax && wave == 0) {      // async fill of next K/V tile
      tdm_load_2d(lds_off(&kbuf[nxt][0][0]),
                  k + base + (size_t)(kb + 1) * 32 * D_, 64, 32, 64, T_, 64);
      tdm_load_2d(lds_off(&vbuf[nxt][0][0]),
                  vT + base + (size_t)(kb + 1) * 32, 32, 64, T_, 64, T_);
    }

    // S = Q K^T for 32 keys: two 16x16 tiles, K-depth 64 = 2 wmma each
    f32x8 s0 = zf, s1 = zf;
#pragma unroll
    for (int kk = 0; kk < 2; ++kk) {
      bf16x16 kf0 = *(const bf16x16*)&kbuf[cur][lm][kk * 32 + half * 16];
      bf16x16 kf1 = *(const bf16x16*)&kbuf[cur][16 + lm][kk * 32 + half * 16];
      s0 = wmma_bf16(qf[kk], kf0, s0);
      s1 = wmma_bf16(qf[kk], kf1, s1);
    }

    // online softmax (fp32); each row is striped over a 16-lane half
    const int ts0 = kb * 32 + lm, ts1 = ts0 + 16;
    const bool need_mask = (kb * 32 + 31) > tq0;
    float p0[8], p1[8], scl[8];
#pragma unroll
    for (int j = 0; j < 8; ++j) {
      int tq = tq0 + j + 8 * half;
      float a = s0[j] * 0.125f;              // 1/sqrt(64)
      float c = s1[j] * 0.125f;
      if (need_mask) {
        if (ts0 > tq) a = NEG;
        if (ts1 > tq) c = NEG;
      }
      float rm = rmax16(fmaxf(a, c));
      float mn = fmaxf(mrow[j], rm);
      float sc = __expf(mrow[j] - mn);
      float e0 = __expf(a - mn);
      float e1 = __expf(c - mn);
      lrow[j] = lrow[j] * sc + rsum16(e0 + e1);
      mrow[j] = mn;
      scl[j] = sc; p0[j] = e0; p1[j] = e1;
    }
#pragma unroll
    for (int dt = 0; dt < 4; ++dt)
#pragma unroll
      for (int j = 0; j < 8; ++j) ao[dt][j] *= scl[j];

    // re-stripe P (C layout) -> A-fragment layout via per-wave LDS
#pragma unroll
    for (int j = 0; j < 8; ++j) {
      int r = j + 8 * half;
      pbuf[wave][r][lm]      = (__bf16)p0[j];
      pbuf[wave][r][16 + lm] = (__bf16)p1[j];
    }
    bf16x16 pf = join8(*(const bf16x8*)&pbuf[wave][lm][b0],
                       *(const bf16x8*)&pbuf[wave][lm][b0 + 8]);
    // O += P(16x32) * V(32x64)
#pragma unroll
    for (int dt = 0; dt < 4; ++dt) {
      bf16x16 vf = *(const bf16x16*)&vbuf[cur][dt * 16 + lm][half * 16];
      ao[dt] = wmma_bf16(pf, vf, ao[dt]);
    }
    if (wave == 0) __builtin_amdgcn_s_wait_tensorcnt(0);
    __syncthreads();
  }

  // normalize and store to (B,T,H*D) bf16 for the output projection
#pragma unroll
  for (int j = 0; j < 8; ++j) {
    float inv = 1.0f / lrow[j];
    int t = tq0 + j + 8 * half;
#pragma unroll
    for (int dt = 0; dt < 4; ++dt) {
      int o = h * D_ + dt * 16 + lm;
      attn[(size_t)(b * T_ + t) * C_ + o] = (__bf16)(ao[dt][j] * inv);
    }
  }
}

// ---------------------------------------------------------------- launch
extern "C" void kernel_launch(void* const* d_in, const int* in_sizes, int n_in,
                              void* d_out, int out_size, void* d_ws, size_t ws_size,
                              hipStream_t stream) {
  const float* x  = (const float*)d_in[0];
  const float* Wk = (const float*)d_in[1];
  const float* Wq = (const float*)d_in[2];
  const float* Wv = (const float*)d_in[3];
  const float* Wp = (const float*)d_in[4];
  float* out = (float*)d_out;

  char* ws = (char*)d_ws;
  size_t off = 0;
  auto alloc = [&](size_t bytes) {
    void* p = ws + off;
    off += (bytes + 255) & ~(size_t)255;
    return p;
  };
  __bf16* xb  = (__bf16*)alloc((size_t)M_ * C_ * 2);
  __bf16* wkb = (__bf16*)alloc((size_t)C_ * C_ * 2);
  __bf16* wqb = (__bf16*)alloc((size_t)C_ * C_ * 2);
  __bf16* wvb = (__bf16*)alloc((size_t)C_ * C_ * 2);
  __bf16* wpb = (__bf16*)alloc((size_t)C_ * C_ * 2);
  __bf16* qd  = (__bf16*)alloc((size_t)M_ * C_ * 2);   // (B,H,T,D)
  __bf16* kd  = (__bf16*)alloc((size_t)M_ * C_ * 2);   // (B,H,T,D)
  __bf16* vtd = (__bf16*)alloc((size_t)M_ * C_ * 2);   // (B,H,D,T)
  __bf16* at  = (__bf16*)alloc((size_t)M_ * C_ * 2);   // (B,T,H*D)

  // fp32 -> bf16
  mha_cvt_bf16<<<(M_ * C_) / 1024, 256, 0, stream>>>(x,  xb,  M_ * C_);
  mha_cvt_bf16<<<(C_ * C_) / 1024, 256, 0, stream>>>(Wk, wkb, C_ * C_);
  mha_cvt_bf16<<<(C_ * C_) / 1024, 256, 0, stream>>>(Wq, wqb, C_ * C_);
  mha_cvt_bf16<<<(C_ * C_) / 1024, 256, 0, stream>>>(Wv, wvb, C_ * C_);
  mha_cvt_bf16<<<(C_ * C_) / 1024, 256, 0, stream>>>(Wp, wpb, C_ * C_);

  // QKV projections (A @ W^T), scattered into attention layouts
  dim3 gg(C_ / 128, M_ / 128);
  mha_gemm_wmma<<<gg, 256, 0, stream>>>(xb, wqb, qd,  M_, C_, C_, 0);
  mha_gemm_wmma<<<gg, 256, 0, stream>>>(xb, wkb, kd,  M_, C_, C_, 0);
  mha_gemm_wmma<<<gg, 256, 0, stream>>>(xb, wvb, vtd, M_, C_, C_, 1);

  // causal flash attention
  mha_flash_wmma<<<dim3(T_ / 64, B_ * H_), 128, 0, stream>>>(qd, kd, vtd, at);

  // output projection, fp32 result
  mha_gemm_wmma<<<gg, 256, 0, stream>>>(at, wpb, out, M_, C_, C_, 2);
}
